// GraphAttentionLayer_29703993819790
// MI455X (gfx1250) — matrix operand order repaired and checked by reading, not compiled, not run on hardware
//
#include <hip/hip_runtime.h>
#include <hip/hip_bf16.h>

typedef __attribute__((ext_vector_type(16))) _Float16 v16h;
typedef __attribute__((ext_vector_type(8)))  float    v8f;

#define BATCH 8
#define NNODE 2048
#define FEAT  128
#define ALPHA 0.2f
#define NEGINF (-9.0e15f)

// A-fragment K-offset for 16-bit 16x32 A layout (contiguous, matches B frags):
// lane half h: elem e -> K = h*16 + e
__device__ __forceinline__ int a_koff(int e, int half) {
    return e + (half << 4);
}

// ---------------------------------------------------------------------------
// Kernel 0: W (f32 [128][128], row=out feat, col=in feat) -> f16 B-fragments.
// Fragment (kc, nt): lane l, elem e -> B[k][n] = W[n][k],
//   n = nt*16 + l%16, k = kc*32 + (l/16)*16 + e
// ---------------------------------------------------------------------------
__global__ __launch_bounds__(256) void wfrag_kernel(const float* __restrict__ W,
                                                    _Float16* __restrict__ wfrag) {
    int t = blockIdx.x * 256 + threadIdx.x;       // 4*8*32*16 = 16384
    int e  = t & 15;
    int l  = (t >> 4) & 31;
    int nt = (t >> 9) & 7;
    int kc = t >> 12;
    int n = nt * 16 + (l & 15);
    int k = kc * 32 + ((l >> 4) << 4) + e;
    wfrag[t] = (_Float16)W[n * FEAT + k];
}

// ---------------------------------------------------------------------------
// Kernel 1: h = input @ W^T via WMMA f16, one wave per 16-row tile.
// Writes h32 [B][N][F] (f32) and hfrag: h pre-swizzled into B-fragment layout
// for the attention GEMM: frag (b, kc2=node/32, nt=feat/16),
//   lane l' = feat%16 + 16*((node%32)/16), elem e' = node%16.
// ---------------------------------------------------------------------------
__global__ __launch_bounds__(32) void h_wmma_kernel(const float* __restrict__ input,
                                                    const _Float16* __restrict__ wfrag,
                                                    float* __restrict__ h32,
                                                    _Float16* __restrict__ hfrag) {
    int blk = blockIdx.x;          // b*128 + mt
    int b  = blk >> 7;
    int mt = blk & 127;
    int L    = threadIdx.x;
    int half = L >> 4;
    int lm   = L & 15;

    // Build A fragments: 16x32 f16 per K-chunk, m = lane%16
    const float* inrow = input + ((size_t)b * NNODE + mt * 16 + lm) * FEAT;
    v16h A[4];
#pragma unroll
    for (int kc = 0; kc < 4; ++kc) {
#pragma unroll
        for (int e = 0; e < 16; ++e)
            A[kc][e] = (_Float16)inrow[kc * 32 + a_koff(e, half)];
    }

    _Float16* hfb = hfrag + (size_t)b * 512 * 32 * 16;
#pragma unroll
    for (int nt = 0; nt < 8; ++nt) {
        v8f acc = {};
#pragma unroll
        for (int kc = 0; kc < 4; ++kc) {
            v16h Bf = *(const v16h*)(wfrag + (((kc * 8 + nt) * 32 + L) << 4));
            acc = __builtin_amdgcn_wmma_f32_16x16x32_f16(
                false, A[kc], false, Bf, (short)0, acc, false, false);
        }
        // D layout: lanes 0-15 VGPR v -> M=v, N=lane; lanes 16-31 -> M=8+v
#pragma unroll
        for (int v = 0; v < 8; ++v) {
            int m    = v + (half << 3);
            int node = mt * 16 + m;
            int feat = nt * 16 + lm;
            float val = acc[v];
            h32[((size_t)b * NNODE + node) * FEAT + feat] = val;
            int fi = (node >> 5) * 8 + nt;
            int lp = lm + ((mt & 1) << 4);
            hfb[(((size_t)fi) * 32 + lp) * 16 + m] = (_Float16)val;
        }
    }
}

// ---------------------------------------------------------------------------
// Kernel 2: src/dst score vectors, one wave per row.
// ---------------------------------------------------------------------------
__global__ __launch_bounds__(32) void score_kernel(const float* __restrict__ h32,
                                                   const float* __restrict__ a_src,
                                                   const float* __restrict__ a_dst,
                                                   float* __restrict__ srcv,
                                                   float* __restrict__ dstv) {
    int row = blockIdx.x;                       // b*N + n
    const float* hrow = h32 + (size_t)row * FEAT;
    float s1 = 0.f, s2 = 0.f;
    for (int f = threadIdx.x; f < FEAT; f += 32) {
        float hv = hrow[f];
        s1 += hv * a_src[f];
        s2 += hv * a_dst[f];
    }
#pragma unroll
    for (int off = 16; off > 0; off >>= 1) {
        s1 += __shfl_xor(s1, off, 32);
        s2 += __shfl_xor(s2, off, 32);
    }
    if (threadIdx.x == 0) { srcv[row] = s1; dstv[row] = s2; }
}

// ---------------------------------------------------------------------------
// Kernel 3: fused masked softmax + P@h (WMMA f16) + ELU.
// One wave per (batch, 16-row tile). adj read exactly twice.
// ---------------------------------------------------------------------------
__global__ __launch_bounds__(32) void attn_kernel(const int* __restrict__ adj,
                                                  const float* __restrict__ srcv,
                                                  const float* __restrict__ dstv,
                                                  const _Float16* __restrict__ hfrag,
                                                  float* __restrict__ out) {
    __shared__ float dst_s[NNODE];
    __shared__ float rmax_s[16];
    __shared__ float rrcp_s[16];

    int blk = blockIdx.x;
    int b  = blk >> 7;
    int mt = blk & 127;
    int rb = mt * 16;
    int L    = threadIdx.x;
    int half = L >> 4;
    int lm   = L & 15;

    const float* dstb = dstv + b * NNODE;
    for (int i = L; i < NNODE; i += 32) dst_s[i] = dstb[i];
    __syncthreads();

    const int* adjb = adj + (size_t)b * NNODE * NNODE;

    // ---- Pass 1: per-row online (max, sum) of masked LeakyReLU scores ----
    for (int r = 0; r < 16; ++r) {
        int row = rb + r;
        float sr = srcv[b * NNODE + row];
        const int* arow = adjb + (size_t)row * NNODE;
        float ml = -3.0e38f, sl = 0.f;
        for (int c = 0; c < 64; ++c) {
            int k = c * 32 + L;
            float ev = sr + dst_s[k];
            ev = ev > 0.f ? ev : ALPHA * ev;
            float x = (arow[k] > 0) ? ev : NEGINF;
            float nm = fmaxf(ml, x);
            sl = sl * __expf(ml - nm) + __expf(x - nm);
            ml = nm;
        }
#pragma unroll
        for (int off = 16; off > 0; off >>= 1) {
            float mo = __shfl_xor(ml, off, 32);
            float so = __shfl_xor(sl, off, 32);
            float nm = fmaxf(ml, mo);
            sl = sl * __expf(ml - nm) + so * __expf(mo - nm);
            ml = nm;
        }
        if (L == 0) { rmax_s[r] = ml; rrcp_s[r] = 1.0f / sl; }
    }
    __syncthreads();

    // ---- Pass 2: P (f16 A-frag) @ h (f16 B-frag) with f32 accumulators ----
    float srcm = srcv[b * NNODE + rb + lm];
    float mrow = rmax_s[lm];
    const int* arow2 = adjb + (size_t)(rb + lm) * NNODE;
    const _Float16* hfb = hfrag + (size_t)b * 512 * 32 * 16;

    v8f acc[8] = {};
    for (int c = 0; c < 64; ++c) {
        int kb = c * 32;
        v16h P;
#pragma unroll
        for (int e = 0; e < 16; ++e) {
            int k = kb + a_koff(e, half);
            float ev = srcm + dst_s[k];
            ev = ev > 0.f ? ev : ALPHA * ev;
            float x = (arow2[k] > 0) ? ev : NEGINF;
            P[e] = (_Float16)__expf(x - mrow);
        }
#pragma unroll
        for (int nt = 0; nt < 8; ++nt) {
            v16h H = *(const v16h*)(hfb + (((size_t)(c * 8 + nt) * 32 + L) << 4));
            acc[nt] = __builtin_amdgcn_wmma_f32_16x16x32_f16(
                false, P, false, H, (short)0, acc[nt], false, false);
        }
    }

    // ---- Epilogue: scale by 1/sum, ELU, store f32 ----
#pragma unroll
    for (int nt = 0; nt < 8; ++nt) {
#pragma unroll
        for (int v = 0; v < 8; ++v) {
            int m = v + (half << 3);
            float val = acc[nt][v] * rrcp_s[m];
            val = val > 0.f ? val : (__expf(val) - 1.0f);
            out[((size_t)b * NNODE + rb + m) * FEAT + nt * 16 + lm] = val;
        }
    }
}

// ---------------------------------------------------------------------------
extern "C" void kernel_launch(void* const* d_in, const int* in_sizes, int n_in,
                              void* d_out, int out_size, void* d_ws, size_t ws_size,
                              hipStream_t stream) {
    const float* input = (const float*)d_in[0];
    const int*   adj   = (const int*)d_in[1];
    const float* W     = (const float*)d_in[2];
    const float* a_src = (const float*)d_in[3];
    const float* a_dst = (const float*)d_in[4];
    float* out = (float*)d_out;

    char* ws = (char*)d_ws;
    float*    h32   = (float*)ws;                                // 8 MB
    _Float16* hfrag = (_Float16*)(ws + 8388608);                 // 4 MB
    _Float16* wfrag = (_Float16*)(ws + 8388608 + 4194304);       // 32 KB
    float*    srcv  = (float*)(ws + 8388608 + 4194304 + 32768);  // 64 KB
    float*    dstv  = (float*)(ws + 8388608 + 4194304 + 32768 + 65536);

    wfrag_kernel<<<64, 256, 0, stream>>>(W, wfrag);
    h_wmma_kernel<<<BATCH * (NNODE / 16), 32, 0, stream>>>(input, wfrag, h32, hfrag);
    score_kernel<<<BATCH * NNODE, 32, 0, stream>>>(h32, a_src, a_dst, srcv, dstv);
    attn_kernel<<<BATCH * (NNODE / 16), 32, 0, stream>>>(adj, srcv, dstv, hfrag, out);
}
